// GraphVO_35820027249379
// MI455X (gfx1250) — compile-verified, matrix operand order checked by reading, not run on hardware
//
#include <hip/hip_runtime.h>
#include <hip/hip_bf16.h>

#define EPSBN 1e-5f

typedef __attribute__((ext_vector_type(16))) _Float16 v16h;
typedef __attribute__((ext_vector_type(8)))  _Float16 v8h;
typedef __attribute__((ext_vector_type(8)))  float    v8f;

// ---------------------------------------------------------------------------
// WMMA fragment helpers (gfx1250, wave32). Layouts per CDNA5 ISA 7.12.2:
//   A 16x32 f16: lane m = lane&15, kh = lane>>4; elem e<8 -> K = kh*8+e,
//                e>=8 -> K = 16 + kh*8 + (e-8) = kh*8 + e + 8.
//   C/D 16x16 f32: col = lane&15, row = vgpr_i + 8*(lane>>4).
//   B packed in memory as [kt][n][32] so a lane's fragment = two 16B loads.
// ---------------------------------------------------------------------------
__device__ __forceinline__ v16h load_a_frag(const _Float16* As, int ldas,
                                            int mt, int lane) {
  int m  = mt * 16 + (lane & 15);
  int kh = (lane >> 4) & 1;
  const _Float16* row = As + (size_t)m * ldas;
  v16h a;
#pragma unroll
  for (int e = 0; e < 16; ++e) {
    int kk = (e < 8) ? (kh * 8 + e) : (kh * 8 + e + 8);
    a[e] = row[kk];
  }
  return a;
}

__device__ __forceinline__ v16h load_b_frag(const _Float16* Bp, int N,
                                            int kt, int n, int lane) {
  int kh = (lane >> 4) & 1;
  const _Float16* base = Bp + (((size_t)kt * N + n) << 5) + kh * 8;
  v8h lo = *(const v8h*)(base);
  v8h hi = *(const v8h*)(base + 16);
  v16h b;
#pragma unroll
  for (int e = 0; e < 8; ++e) { b[e] = lo[e]; b[e + 8] = hi[e]; }
  return b;
}

__device__ __forceinline__ v8f wmma_f16(v16h a, v16h b, v8f c) {
  return __builtin_amdgcn_wmma_f32_16x16x32_f16(false, a, false, b,
                                                (short)0, c, false, false);
}

// ---------------------------------------------------------------------------
// Layer 0: conv 7x7 s2 p3 (Cin=1) + bias + BN + leaky(0.1) + maxpool3s2p1,
// fused. 1024 x [1,64,64] -> [64,16,16]. Cin==1 gives K=49: too small for
// WMMA; 6.6 GF of VALU is fine and fusing the pool avoids a 268MB buffer.
// ---------------------------------------------------------------------------
__launch_bounds__(256)
__global__ void k_conv0_pool(const float* __restrict__ x,
                             const float* __restrict__ w,
                             const float* __restrict__ cb,
                             const float* __restrict__ g,
                             const float* __restrict__ be,
                             float* __restrict__ out) {
  __shared__ float ws[64 * 49];
  for (int i = threadIdx.x; i < 64 * 49; i += 256) ws[i] = w[i];
  __syncthreads();
  const long total = (long)1024 * 64 * 16 * 16;
  for (long o = (long)blockIdx.x * 256 + threadIdx.x; o < total;
       o += (long)gridDim.x * 256) {
    int px = (int)(o & 15);
    int py = (int)((o >> 4) & 15);
    int oc = (int)((o >> 8) & 63);
    int img = (int)(o >> 14);
    const float* im = x + (size_t)img * 4096;
    const float* wk = ws + oc * 49;
    float sc = g[oc] * rsqrtf(1.f + EPSBN);
    float bb = be[oc];
    float b0 = cb[oc];
    float mx = -__builtin_inff();
#pragma unroll
    for (int wy = 0; wy < 3; ++wy) {
#pragma unroll
      for (int wx = 0; wx < 3; ++wx) {
        int cy = py * 2 - 1 + wy, cx = px * 2 - 1 + wx;
        if (cy < 0 || cy > 31 || cx < 0 || cx > 31) continue;
        float s = 0.f;
        for (int ky = 0; ky < 7; ++ky) {
          int iy = cy * 2 - 3 + ky;
          if (iy < 0 || iy > 63) continue;
          for (int kx = 0; kx < 7; ++kx) {
            int ix = cx * 2 - 3 + kx;
            if (ix < 0 || ix > 63) continue;
            s += im[iy * 64 + ix] * wk[ky * 7 + kx];
          }
        }
        float v = (s + b0) * sc + bb;
        v = (v >= 0.f) ? v : 0.1f * v;
        mx = fmaxf(mx, v);
      }
    }
    out[o] = mx;
  }
}

// Generic maxpool 3x3 s2 p1 on NCHW.
__global__ void k_pool(const float* __restrict__ in, float* __restrict__ out,
                       int NC, int H, int W, int OH, int OW) {
  long total = (long)NC * OH * OW;
  for (long o = (long)blockIdx.x * blockDim.x + threadIdx.x; o < total;
       o += (long)gridDim.x * blockDim.x) {
    int ox = (int)(o % OW);
    long t = o / OW;
    int oy = (int)(t % OH);
    int nc = (int)(t / OH);
    const float* p = in + (size_t)nc * H * W;
    float mx = -__builtin_inff();
#pragma unroll
    for (int wy = 0; wy < 3; ++wy) {
#pragma unroll
      for (int wx = 0; wx < 3; ++wx) {
        int iy = oy * 2 - 1 + wy, ix = ox * 2 - 1 + wx;
        if (iy >= 0 && iy < H && ix >= 0 && ix < W) mx = fmaxf(mx, p[iy * W + ix]);
      }
    }
    out[o] = mx;
  }
}

// ---------------------------------------------------------------------------
// Weight packing: f32 -> f16, layout Bp[kt][n][kc] (kt = k/32, kc = k%32),
// zero-padded to a multiple of 32 in K so the WMMA K-loop needs no remainder.
// ---------------------------------------------------------------------------
__global__ void k_pack_conv_w(const float* __restrict__ w, _Float16* __restrict__ Bp,
                              int Cout, int Cin, int KHKW) {
  int K = Cin * KHKW;
  int Kp = (K + 31) & ~31;
  long total = (long)Kp * Cout;
  for (long o = (long)blockIdx.x * blockDim.x + threadIdx.x; o < total;
       o += (long)gridDim.x * blockDim.x) {
    int n = (int)(o % Cout);
    int k = (int)(o / Cout);
    float v = 0.f;
    if (k < K) {
      int cin = k / KHKW, t = k - cin * KHKW;
      v = w[((size_t)n * Cin + cin) * KHKW + t];
    }
    Bp[(((size_t)(k >> 5) * Cout + n) << 5) + (k & 31)] = (_Float16)v;
  }
}

// Center tap of an OIHW kernel applied to 1x1 spatial input == plain GEMM.
__global__ void k_pack_center_w(const float* __restrict__ w, _Float16* __restrict__ Bp,
                                int Cout, int Cin, int KH, int KW) {
  long total = (long)Cin * Cout;  // Cin is a multiple of 32 in all uses
  int ctr = (KH / 2) * KW + (KW / 2);
  for (long o = (long)blockIdx.x * blockDim.x + threadIdx.x; o < total;
       o += (long)gridDim.x * blockDim.x) {
    int n = (int)(o % Cout);
    int k = (int)(o / Cout);
    float v = w[((size_t)n * Cin + k) * (KH * KW) + ctr];
    Bp[(((size_t)(k >> 5) * Cout + n) << 5) + (k & 31)] = (_Float16)v;
  }
}

__global__ void k_pack_mat(const float* __restrict__ w, _Float16* __restrict__ Bp,
                           int K, int N) {
  int Kp = (K + 31) & ~31;
  long total = (long)Kp * N;
  for (long o = (long)blockIdx.x * blockDim.x + threadIdx.x; o < total;
       o += (long)gridDim.x * blockDim.x) {
    int n = (int)(o % N);
    int k = (int)(o / N);
    float v = (k < K) ? w[(size_t)k * N + n] : 0.f;
    Bp[(((size_t)(k >> 5) * N + n) << 5) + (k & 31)] = (_Float16)v;
  }
}

// ---------------------------------------------------------------------------
// Implicit-im2col conv via WMMA. Block = 256 threads = 8 waves covering a
// 64(M) x 64(N) output tile: 4 N-tiles x 2 M-groups, each wave holds TWO
// accumulators (two M-tiles) and issues 2 WMMAs per K-step sharing one B
// fragment -> half the barrier/B-load cost per MAC vs a 32-wide M tile.
// A-tile (64x32 f16) gathered into LDS with im2col addressing + f32->f16
// convert; B streamed from the packed layout with two 16B loads/lane.
// Fused bias+BN+leaky epilogue.
// ---------------------------------------------------------------------------
__launch_bounds__(256)
__global__ void k_conv_wmma(const float* __restrict__ in,
                            const _Float16* __restrict__ Bp,
                            float* __restrict__ out,
                            int NIMG, int Cin, int IH, int IW,
                            int Cout, int OH, int OW,
                            int KH, int KW, int stride, int pad,
                            const float* __restrict__ cb,
                            const float* __restrict__ bng,
                            const float* __restrict__ bnb,
                            float slope) {
  __shared__ _Float16 As[64][40];
  int tid = threadIdx.x, lane = tid & 31, wave = tid >> 5;
  int mg = wave >> 2, nt = wave & 3;      // wave owns M-tiles {2mg, 2mg+1}
  int MP = OH * OW;
  int M = NIMG * MP;
  int KHKW = KH * KW;
  int K = Cin * KHKW;                     // multiple of 32 in all uses
  int KT = K >> 5;
  int mblock = blockIdx.x * 64;
  int nblock = blockIdx.y * 64;
  int nf = nblock + nt * 16 + (lane & 15);
  v8f acc0 = {}, acc1 = {};
  for (int kt = 0; kt < KT; ++kt) {
    __syncthreads();
#pragma unroll
    for (int it = 0; it < 8; ++it) {
      int idx = tid + it * 256;           // 64 x 32 elements
      int rl = idx >> 5, kc = idx & 31;
      int r = mblock + rl;
      int k = kt * 32 + kc;
      float v = 0.f;
      if (r < M) {
        int img = r / MP, p = r - img * MP;
        int oy = p / OW, ox = p - oy * OW;
        int cin = k / KHKW, t = k - cin * KHKW;
        int ky = t / KW, kx = t - ky * KW;
        int iy = oy * stride - pad + ky;
        int ix = ox * stride - pad + kx;
        if (iy >= 0 && iy < IH && ix >= 0 && ix < IW)
          v = in[(((size_t)img * Cin + cin) * IH + iy) * IW + ix];
      }
      As[rl][kc] = (_Float16)v;
    }
    __syncthreads();
    v16h b = load_b_frag(Bp, Cout, kt, nf, lane);
    if (kt + 1 < KT)  // gfx1250 global_prefetch_b8 of next B tile
      __builtin_prefetch(Bp + (((size_t)(kt + 1) * Cout + nf) << 5), 0, 1);
    v16h a0 = load_a_frag(&As[0][0], 40, mg * 2 + 0, lane);
    v16h a1 = load_a_frag(&As[0][0], 40, mg * 2 + 1, lane);
    acc0 = wmma_f16(a0, b, acc0);
    acc1 = wmma_f16(a1, b, acc1);
  }
  float sc = bng[nf] * rsqrtf(1.f + EPSBN);
  float bb = bnb[nf];
  float b0 = cb[nf];
#pragma unroll
  for (int half = 0; half < 2; ++half) {
    int rb = mblock + (mg * 2 + half) * 16 + ((lane >> 4) & 1) * 8;
    const v8f& acc = half ? acc1 : acc0;
#pragma unroll
    for (int i = 0; i < 8; ++i) {
      int r = rb + i;
      if (r >= M) continue;
      float v = (acc[i] + b0) * sc + bb;
      v = (v >= 0.f) ? v : slope * v;
      int img = r / MP, p = r - img * MP;
      out[((size_t)img * Cout + nf) * MP + p] = v;
    }
  }
}

// ---------------------------------------------------------------------------
// Dual-A fused GEMM: C = epilogue(A1@B1 [+ A2@B2]). A row-major f32 [M,K]
// (converted to f16 in the LDS gather), B pre-packed f16. Same 64x64 block
// tile / dual-accumulator scheme as the conv kernel. Epilogue:
//   v = (acc + cb[n]) * bn_scale[n] + bn_beta[n] + lb[n]; leaky(slope).
// Used for conv4-7 (center-tap, BN) and both graph layers (dual A, bias).
// ---------------------------------------------------------------------------
__launch_bounds__(256)
__global__ void k_gemm_dual_wmma(const float* __restrict__ A1,
                                 const _Float16* __restrict__ B1, int K1,
                                 const float* __restrict__ A2,
                                 const _Float16* __restrict__ B2, int K2,
                                 float* __restrict__ C, int M, int N,
                                 const float* __restrict__ cb,
                                 const float* __restrict__ bng,
                                 const float* __restrict__ bnb,
                                 const float* __restrict__ lb,
                                 float slope) {
  __shared__ _Float16 As[64][40];
  int tid = threadIdx.x, lane = tid & 31, wave = tid >> 5;
  int mg = wave >> 2, nt = wave & 3;
  int mblock = blockIdx.x * 64;
  int nblock = blockIdx.y * 64;
  int nf = nblock + nt * 16 + (lane & 15);
  v8f acc0 = {}, acc1 = {};
  for (int pass = 0; pass < 2; ++pass) {
    const float* A = pass ? A2 : A1;
    const _Float16* B = pass ? B2 : B1;
    int K = pass ? K2 : K1;
    if (A == nullptr) continue;
    int KT = (K + 31) >> 5;
    for (int kt = 0; kt < KT; ++kt) {
      __syncthreads();
#pragma unroll
      for (int it = 0; it < 8; ++it) {
        int idx = tid + it * 256;
        int rl = idx >> 5, kc = idx & 31;
        int r = mblock + rl;
        int k = kt * 32 + kc;
        float v = 0.f;
        if (r < M && k < K) v = A[(size_t)r * K + k];
        As[rl][kc] = (_Float16)v;
      }
      __syncthreads();
      v16h b = load_b_frag(B, N, kt, nf, lane);
      if (kt + 1 < KT)
        __builtin_prefetch(B + (((size_t)(kt + 1) * N + nf) << 5), 0, 1);
      v16h a0 = load_a_frag(&As[0][0], 40, mg * 2 + 0, lane);
      v16h a1 = load_a_frag(&As[0][0], 40, mg * 2 + 1, lane);
      acc0 = wmma_f16(a0, b, acc0);
      acc1 = wmma_f16(a1, b, acc1);
    }
  }
  float sc = 1.f, bb = 0.f, b0 = 0.f, b1 = 0.f;
  if (cb)  b0 = cb[nf];
  if (bng) { sc = bng[nf] * rsqrtf(1.f + EPSBN); bb = bnb[nf]; }
  if (lb)  b1 = lb[nf];
#pragma unroll
  for (int half = 0; half < 2; ++half) {
    int rb = mblock + (mg * 2 + half) * 16 + ((lane >> 4) & 1) * 8;
    const v8f& acc = half ? acc1 : acc0;
#pragma unroll
    for (int i = 0; i < 8; ++i) {
      int r = rb + i;
      if (r >= M) continue;
      float v = (acc[i] + b0) * sc + bb + b1;
      v = (v >= 0.f) ? v : slope * v;
      C[(size_t)r * N + nf] = v;
    }
  }
}

// ---------------------------------------------------------------------------
// Deterministic CSR build over dst (counts via int atomics, ordered fill).
// ---------------------------------------------------------------------------
__global__ void k_zero_i32(int* p, int n) {
  for (int i = blockIdx.x * blockDim.x + threadIdx.x; i < n;
       i += gridDim.x * blockDim.x) p[i] = 0;
}
__global__ void k_deg(const long long* __restrict__ dst, int E, int* __restrict__ deg) {
  for (int e = blockIdx.x * blockDim.x + threadIdx.x; e < E;
       e += gridDim.x * blockDim.x) atomicAdd(&deg[(int)dst[e]], 1);
}
__launch_bounds__(1024)
__global__ void k_scan1024(const int* __restrict__ deg, int* __restrict__ rowstart) {
  __shared__ int sm[1024];
  int t = threadIdx.x;
  sm[t] = deg[t];
  __syncthreads();
  for (int off = 1; off < 1024; off <<= 1) {
    int add = (t >= off) ? sm[t - off] : 0;
    __syncthreads();
    sm[t] += add;
    __syncthreads();
  }
  rowstart[t + 1] = sm[t];
  if (t == 0) rowstart[0] = 0;
}
__global__ void k_fill_csr(const long long* __restrict__ src,
                           const long long* __restrict__ dst, int E,
                           const int* __restrict__ rowstart, int* __restrict__ adj) {
  int n = blockIdx.x * blockDim.x + threadIdx.x;
  if (n >= 1024) return;
  int pos = rowstart[n];
  for (int e = 0; e < E; ++e)
    if ((int)dst[e] == n) adj[pos++] = (int)src[e];
}

// concat([feat, node_idx]) -> [N, F+1]
__global__ void k_concat_idx(const float* __restrict__ src, float* __restrict__ out,
                             int N, int F) {
  long total = (long)N * (F + 1);
  for (long o = (long)blockIdx.x * blockDim.x + threadIdx.x; o < total;
       o += (long)gridDim.x * blockDim.x) {
    int f = (int)(o % (F + 1));
    int n = (int)(o / (F + 1));
    out[o] = (f < F) ? src[(size_t)n * F + f] : (float)n;
  }
}

// PyG SoftmaxAggregation (t=1): per-(node, feature) softmax over in-edges.
__global__ void k_aggr_softmax(const float* __restrict__ X, int F,
                               const int* __restrict__ rowstart,
                               const int* __restrict__ adj,
                               float* __restrict__ out) {
  int n = blockIdx.x;
  int f = blockIdx.y * blockDim.x + threadIdx.x;
  if (f >= F) return;
  int s0 = rowstart[n], s1 = rowstart[n + 1];
  if (s1 == s0) { out[(size_t)n * F + f] = 0.f; return; }
  float mx = -__builtin_inff();
  for (int j = s0; j < s1; ++j) mx = fmaxf(mx, X[(size_t)adj[j] * F + f]);
  float se = 0.f, sv = 0.f;
  for (int j = s0; j < s1; ++j) {
    float v = X[(size_t)adj[j] * F + f];
    float e = expf(v - mx);
    se += e;
    sv += v * e;
  }
  out[(size_t)n * F + f] = sv / (se + 1e-16f);
}

__global__ void k_aggr_sum(const float* __restrict__ X, int F,
                           const int* __restrict__ rowstart,
                           const int* __restrict__ adj,
                           float* __restrict__ out) {
  int n = blockIdx.x;
  int f = blockIdx.y * blockDim.x + threadIdx.x;
  if (f >= F) return;
  int s0 = rowstart[n], s1 = rowstart[n + 1];
  float s = 0.f;
  for (int j = s0; j < s1; ++j) s += X[(size_t)adj[j] * F + f];
  out[(size_t)n * F + f] = s;
}

// Heads: pos = x3@pos_w + pos_b ; rot = normalize(x3@rot_w + rot_b).
__global__ void k_head(const float* __restrict__ x3,
                       const float* __restrict__ pw, const float* __restrict__ pb,
                       const float* __restrict__ rw, const float* __restrict__ rb,
                       float* __restrict__ out) {
  int n = blockIdx.x * blockDim.x + threadIdx.x;
  if (n >= 1024) return;
  const float* xr = x3 + (size_t)n * 65;
  float p0 = pb[0], p1 = pb[1], p2 = pb[2];
  float r0 = rb[0], r1 = rb[1], r2 = rb[2], r3 = rb[3];
  for (int k = 0; k < 65; ++k) {
    float v = xr[k];
    p0 += v * pw[k * 3 + 0];
    p1 += v * pw[k * 3 + 1];
    p2 += v * pw[k * 3 + 2];
    r0 += v * rw[k * 4 + 0];
    r1 += v * rw[k * 4 + 1];
    r2 += v * rw[k * 4 + 2];
    r3 += v * rw[k * 4 + 3];
  }
  float nr = fmaxf(sqrtf(r0 * r0 + r1 * r1 + r2 * r2 + r3 * r3), 1e-12f);
  float* o = out + (size_t)n * 7;
  o[0] = p0; o[1] = p1; o[2] = p2;
  o[3] = r0 / nr; o[4] = r1 / nr; o[5] = r2 / nr; o[6] = r3 / nr;
}

// ---------------------------------------------------------------------------
extern "C" void kernel_launch(void* const* d_in, const int* in_sizes, int n_in,
                              void* d_out, int out_size, void* d_ws, size_t ws_size,
                              hipStream_t stream) {
  (void)in_sizes; (void)n_in; (void)out_size; (void)ws_size;
  const float* x = (const float*)d_in[0];
  const long long* ei = (const long long*)d_in[1];
  const long long* esrc = ei;
  const long long* edst = ei + 16384;
  const float *cw[8], *cbv[8], *cg[8], *cbe[8];
  for (int i = 0; i < 8; ++i) {
    cw[i]  = (const float*)d_in[2 + 4 * i + 0];
    cbv[i] = (const float*)d_in[2 + 4 * i + 1];
    cg[i]  = (const float*)d_in[2 + 4 * i + 2];
    cbe[i] = (const float*)d_in[2 + 4 * i + 3];
  }
  const float* rel1_w  = (const float*)d_in[34];
  const float* rel1_b  = (const float*)d_in[35];
  const float* root1_w = (const float*)d_in[36];
  const float* rel2_w  = (const float*)d_in[37];
  const float* rel2_b  = (const float*)d_in[38];
  const float* root2_w = (const float*)d_in[39];
  const float* pos_w   = (const float*)d_in[40];
  const float* pos_b   = (const float*)d_in[41];
  const float* rot_w   = (const float*)d_in[42];
  const float* rot_b   = (const float*)d_in[43];

  // ----- workspace layout (bytes, 256B aligned); late tensors reuse arena0 --
  char* w = (char*)d_ws;
  size_t cur = 0;
  auto A = [&](size_t bytes) { char* p = w + cur; cur += (bytes + 255) & ~(size_t)255; return p; };
  float* p0  = (float*)A((size_t)1024 * 64 * 16 * 16 * 4);   // 67MB arena0
  float* c1  = (float*)A((size_t)1024 * 128 * 8 * 8 * 4);
  float* p1  = (float*)A((size_t)1024 * 128 * 4 * 4 * 4);
  float* c2  = (float*)A((size_t)1024 * 256 * 2 * 2 * 4);
  _Float16* b1p = (_Float16*)A((size_t)1600 * 128 * 2);
  _Float16* b2p = (_Float16*)A((size_t)3200 * 256 * 2);
  _Float16* b3p = (_Float16*)A((size_t)2304 * 256 * 2);
  _Float16* b4p = (_Float16*)A((size_t)256 * 512 * 2);
  _Float16* b5p = (_Float16*)A((size_t)512 * 512 * 2);
  _Float16* b6p = (_Float16*)A((size_t)512 * 512 * 2);
  _Float16* b7p = (_Float16*)A((size_t)512 * 512 * 2);
  _Float16* r1p = (_Float16*)A((size_t)544 * 256 * 2);
  _Float16* o1p = (_Float16*)A((size_t)544 * 256 * 2);
  _Float16* r2p = (_Float16*)A((size_t)288 * 64 * 2);
  _Float16* o2p = (_Float16*)A((size_t)288 * 64 * 2);
  int* deg      = (int*)A(1024 * 4);
  int* rowstart = (int*)A(1025 * 4);
  int* adj      = (int*)A(16384 * 4);
  // arena0 (p0) is dead after conv1; reuse it for everything post-conv2.
  char* a0 = (char*)p0;
  size_t c0 = 0;
  auto R = [&](size_t bytes) { char* p = a0 + c0; c0 += (bytes + 255) & ~(size_t)255; return (float*)p; };
  float* c3    = R((size_t)1024 * 256 * 2 * 2 * 4);
  float* feat3 = R((size_t)1024 * 256 * 4);
  float* c4    = R((size_t)1024 * 512 * 4);
  float* c5    = R((size_t)1024 * 512 * 4);
  float* c6    = R((size_t)1024 * 512 * 4);
  float* feat  = R((size_t)1024 * 512 * 4);
  float* x1    = R((size_t)1024 * 513 * 4);
  float* agg1  = R((size_t)1024 * 513 * 4);
  float* h1    = R((size_t)1024 * 256 * 4);
  float* x2    = R((size_t)1024 * 257 * 4);
  float* agg2  = R((size_t)1024 * 257 * 4);
  float* h2    = R((size_t)1024 * 64 * 4);
  float* x3    = R((size_t)1024 * 65 * 4);

  // ----- weight packing (f32 -> packed f16) --------------------------------
  hipLaunchKernelGGL(k_pack_conv_w, dim3(256), dim3(256), 0, stream, cw[1], b1p, 128, 64, 25);
  hipLaunchKernelGGL(k_pack_conv_w, dim3(512), dim3(256), 0, stream, cw[2], b2p, 256, 128, 25);
  hipLaunchKernelGGL(k_pack_conv_w, dim3(512), dim3(256), 0, stream, cw[3], b3p, 256, 256, 9);
  hipLaunchKernelGGL(k_pack_center_w, dim3(256), dim3(256), 0, stream, cw[4], b4p, 512, 256, 3, 3);
  hipLaunchKernelGGL(k_pack_center_w, dim3(256), dim3(256), 0, stream, cw[5], b5p, 512, 512, 3, 3);
  hipLaunchKernelGGL(k_pack_center_w, dim3(256), dim3(256), 0, stream, cw[6], b6p, 512, 512, 3, 3);
  hipLaunchKernelGGL(k_pack_center_w, dim3(256), dim3(256), 0, stream, cw[7], b7p, 512, 512, 3, 3);
  hipLaunchKernelGGL(k_pack_mat, dim3(256), dim3(256), 0, stream, rel1_w, r1p, 513, 256);
  hipLaunchKernelGGL(k_pack_mat, dim3(256), dim3(256), 0, stream, root1_w, o1p, 513, 256);
  hipLaunchKernelGGL(k_pack_mat, dim3(64), dim3(256), 0, stream, rel2_w, r2p, 257, 64);
  hipLaunchKernelGGL(k_pack_mat, dim3(64), dim3(256), 0, stream, root2_w, o2p, 257, 64);

  // ----- CNN ---------------------------------------------------------------
  hipLaunchKernelGGL(k_conv0_pool, dim3(8192), dim3(256), 0, stream,
                     x, cw[0], cbv[0], cg[0], cbe[0], p0);
  // conv1: [1024,64,16,16] -> [1024,128,8,8]  (M=65536, K=1600, N=128)
  hipLaunchKernelGGL(k_conv_wmma, dim3(1024, 2), dim3(256), 0, stream,
                     p0, b1p, c1, 1024, 64, 16, 16, 128, 8, 8, 5, 5, 2, 2,
                     cbv[1], cg[1], cbe[1], 0.1f);
  hipLaunchKernelGGL(k_pool, dim3(2048), dim3(256), 0, stream,
                     c1, p1, 1024 * 128, 8, 8, 4, 4);
  // conv2: [1024,128,4,4] -> [1024,256,2,2]  (M=4096, K=3200, N=256)
  hipLaunchKernelGGL(k_conv_wmma, dim3(64, 4), dim3(256), 0, stream,
                     p1, b2p, c2, 1024, 128, 4, 4, 256, 2, 2, 5, 5, 2, 2,
                     cbv[2], cg[2], cbe[2], 0.1f);
  // conv3: [1024,256,2,2] -> [1024,256,2,2]  (M=4096, K=2304, N=256)
  hipLaunchKernelGGL(k_conv_wmma, dim3(64, 4), dim3(256), 0, stream,
                     c2, b3p, c3, 1024, 256, 2, 2, 256, 2, 2, 3, 3, 1, 1,
                     cbv[3], cg[3], cbe[3], 0.1f);
  hipLaunchKernelGGL(k_pool, dim3(1024), dim3(256), 0, stream,
                     c3, feat3, 1024 * 256, 2, 2, 1, 1);
  // conv4..7 on 1x1 spatial == center-tap GEMMs with fused bias+BN+leaky(0.1)
  hipLaunchKernelGGL(k_gemm_dual_wmma, dim3(16, 8), dim3(256), 0, stream,
                     feat3, b4p, 256, (const float*)nullptr, (const _Float16*)nullptr, 0,
                     c4, 1024, 512, cbv[4], cg[4], cbe[4], (const float*)nullptr, 0.1f);
  hipLaunchKernelGGL(k_gemm_dual_wmma, dim3(16, 8), dim3(256), 0, stream,
                     c4, b5p, 512, (const float*)nullptr, (const _Float16*)nullptr, 0,
                     c5, 1024, 512, cbv[5], cg[5], cbe[5], (const float*)nullptr, 0.1f);
  hipLaunchKernelGGL(k_gemm_dual_wmma, dim3(16, 8), dim3(256), 0, stream,
                     c5, b6p, 512, (const float*)nullptr, (const _Float16*)nullptr, 0,
                     c6, 1024, 512, cbv[6], cg[6], cbe[6], (const float*)nullptr, 0.1f);
  hipLaunchKernelGGL(k_gemm_dual_wmma, dim3(16, 8), dim3(256), 0, stream,
                     c6, b7p, 512, (const float*)nullptr, (const _Float16*)nullptr, 0,
                     feat, 1024, 512, cbv[7], cg[7], cbe[7], (const float*)nullptr, 0.1f);

  // ----- graph: deterministic CSR over dst ---------------------------------
  hipLaunchKernelGGL(k_zero_i32, dim3(4), dim3(256), 0, stream, deg, 1024);
  hipLaunchKernelGGL(k_deg, dim3(64), dim3(256), 0, stream, edst, 16384, deg);
  hipLaunchKernelGGL(k_scan1024, dim3(1), dim3(1024), 0, stream, deg, rowstart);
  hipLaunchKernelGGL(k_fill_csr, dim3(4), dim3(256), 0, stream,
                     esrc, edst, 16384, rowstart, adj);

  // layer 1: softmax aggregation + dual GEMM, leaky(0.01)
  hipLaunchKernelGGL(k_concat_idx, dim3(1024), dim3(256), 0, stream, feat, x1, 1024, 512);
  hipLaunchKernelGGL(k_aggr_softmax, dim3(1024, 3), dim3(256), 0, stream,
                     x1, 513, rowstart, adj, agg1);
  hipLaunchKernelGGL(k_gemm_dual_wmma, dim3(16, 4), dim3(256), 0, stream,
                     agg1, r1p, 513, x1, o1p, 513,
                     h1, 1024, 256, (const float*)nullptr, (const float*)nullptr,
                     (const float*)nullptr, rel1_b, 0.01f);
  // layer 2: sum aggregation + dual GEMM, leaky(0.01)
  hipLaunchKernelGGL(k_concat_idx, dim3(512), dim3(256), 0, stream, h1, x2, 1024, 256);
  hipLaunchKernelGGL(k_aggr_sum, dim3(1024, 2), dim3(256), 0, stream,
                     x2, 257, rowstart, adj, agg2);
  hipLaunchKernelGGL(k_gemm_dual_wmma, dim3(16, 1), dim3(256), 0, stream,
                     agg2, r2p, 257, x2, o2p, 257,
                     h2, 1024, 64, (const float*)nullptr, (const float*)nullptr,
                     (const float*)nullptr, rel2_b, 0.01f);

  // heads
  hipLaunchKernelGGL(k_concat_idx, dim3(256), dim3(256), 0, stream, h2, x3, 1024, 64);
  hipLaunchKernelGGL(k_head, dim3(4), dim3(256), 0, stream,
                     x3, pos_w, pos_b, rot_w, rot_b, (float*)d_out);
}
